// CrossAttentionLayer_11012296147633
// MI455X (gfx1250) — matrix operand order repaired and checked by reading, compiled, not run on hardware
//
#include <hip/hip_runtime.h>
#include <cstdint>

// ---------------------------------------------------------------------------
// CrossAttentionLayer for MI455X (gfx1250).
// All GEMM/attention operands staged once as f16; WMMA f16 with fp32 accum.
// fp32 kept for: accumulators, softmax, residual+LayerNorm, all d_out writes.
// Inner loops are software-pipelined one full K-step ahead so every WMMA
// consumes fragments whose loads were issued an entire iteration earlier.
// D=512, H=8, hd=64, B=8, L=S=1024  ->  M (tokens) = 8192.
// ---------------------------------------------------------------------------

typedef __attribute__((ext_vector_type(16))) _Float16 v16h;
typedef __attribute__((ext_vector_type(8)))  _Float16 v8h;
typedef __attribute__((ext_vector_type(16))) float    v16f;
typedef __attribute__((ext_vector_type(8)))  float    v8f;

#define DMODEL 512
#define NHEAD  8
#define HDIM   64
#define SEQ    1024
#define NTOK   8192

static __device__ __forceinline__ float wred_max(float v) {
#pragma unroll
  for (int o = 16; o > 0; o >>= 1) v = fmaxf(v, __shfl_xor(v, o, 32));
  return v;
}
static __device__ __forceinline__ float wred_sum(float v) {
#pragma unroll
  for (int o = 16; o > 0; o >>= 1) v += __shfl_xor(v, o, 32);
  return v;
}
static __device__ __forceinline__ float4 ld4(const float* p) {
  return *(const float4*)p;
}
// Pack 4 float4 chunks (16 fp32) into a 16xf16 WMMA fragment (LDS P-frag).
static __device__ __forceinline__ v16h pack16(const float4& x0, const float4& x1,
                                              const float4& x2, const float4& x3) {
  v16f f;
  f[0] = x0.x;  f[1] = x0.y;  f[2] = x0.z;  f[3] = x0.w;
  f[4] = x1.x;  f[5] = x1.y;  f[6] = x1.z;  f[7] = x1.w;
  f[8] = x2.x;  f[9] = x2.y;  f[10] = x2.z; f[11] = x2.w;
  f[12] = x3.x; f[13] = x3.y; f[14] = x3.z; f[15] = x3.w;
  return __builtin_convertvector(f, v16h);
}
static __device__ __forceinline__ v16h cat8(const v8h& lo, const v8h& hi) {
  return __builtin_shufflevector(lo, hi, 0, 1, 2, 3, 4, 5, 6, 7,
                                 8, 9, 10, 11, 12, 13, 14, 15);
}

// ---------------------------------------------------------------------------
// fp32 -> f16 streaming convert (8 elements / thread; n % 8 == 0).
// ---------------------------------------------------------------------------
__global__ __launch_bounds__(256) void cvt_kernel(const float* __restrict__ s,
                                                  _Float16* __restrict__ d,
                                                  int n) {
  const int i = (blockIdx.x * 256 + threadIdx.x) * 8;
  if (i >= n) return;
  const float4 a = ld4(s + i), b = ld4(s + i + 4);
  v8f f;
  f[0] = a.x; f[1] = a.y; f[2] = a.z; f[3] = a.w;
  f[4] = b.x; f[5] = b.y; f[6] = b.z; f[7] = b.w;
  *(v8h*)(d + i) = __builtin_convertvector(f, v8h);
}

// ---------------------------------------------------------------------------
// Generic GEMM:  C[m,n] = relu?( sum_k A[m,k] * W[n,k] + bias[n] )
// A: (M,K) f16 row-major.  W: (N,K) f16 row-major (computes A @ W^T).
// outF: optional fp32 result; outH: optional f16 result.
// flags: bit0 = ReLU, bit1 = f16 store in transposed-V layout (B,H,hd,S).
// Block = 256 threads (8 waves). Wave tile = 16(M) x 64(N). Block = 128M x 64N.
// K is a template parameter (512 / 1024, power of two) so the software
// pipeline can unroll + register-rotate.
// ---------------------------------------------------------------------------
template <int K>
__global__ __launch_bounds__(256) void gemm_kernel(
    const _Float16* __restrict__ A, const _Float16* __restrict__ W,
    const float* __restrict__ bias, float* __restrict__ outF,
    _Float16* __restrict__ outH, int N, int flags) {
  const int lane = threadIdx.x & 31;
  const int wave = threadIdx.x >> 5;
  const int g = lane >> 4;      // half-wave group (0/1)
  const int r = lane & 15;      // row (A) / row-of-W (B) within tile
  const int m0 = blockIdx.y * 128 + wave * 16;
  const int n0 = blockIdx.x * 64;

  v8f acc[4] = {};
  const _Float16* arow = A + (size_t)(m0 + r) * K + 16 * g;
  const _Float16* wrow0 = W + (size_t)(n0 + r) * K + 16 * g;
  constexpr size_t wstride = (size_t)16 * K;   // 16 rows of W per n-tile

  // A frag: lane g*16+r holds A[m0+r, kk+16g+i], i = 0..15 (contiguous)
  // B frag: lane g*16+r holds W[n0+16t+r, kk+16g+i], i = 0..15 (contiguous)
  v16h a_cur = cat8(*(const v8h*)(arow), *(const v8h*)(arow + 8));
  v16h b_cur[4];
#pragma unroll
  for (int t = 0; t < 4; ++t)
    b_cur[t] = *(const v16h*)(wrow0 + t * wstride);

#pragma unroll 4
  for (int kk = 0; kk < K; kk += 32) {
    const int kn = (kk + 32) & (K - 1);   // wrap on last step (harmless reload)
    // issue next K-step's 10 b128 loads before consuming current fragments
    const v16h a_nxt =
        cat8(*(const v8h*)(arow + kn), *(const v8h*)(arow + kn + 8));
    v16h b_nxt[4];
#pragma unroll
    for (int t = 0; t < 4; ++t)
      b_nxt[t] = *(const v16h*)(wrow0 + t * wstride + kn);

#pragma unroll
    for (int t = 0; t < 4; ++t)
      acc[t] = __builtin_amdgcn_wmma_f32_16x16x32_f16(
          false, a_cur, false, b_cur[t], (short)0, acc[t], false, false);

    a_cur = a_nxt;
#pragma unroll
    for (int t = 0; t < 4; ++t) b_cur[t] = b_nxt[t];
  }

  // Epilogue: C/D layout -> VGPR e, lane g*16+n == element (M=8g+e, N=n)
#pragma unroll
  for (int t = 0; t < 4; ++t) {
    const int n = n0 + 16 * t + r;
    const float bv = bias ? bias[n] : 0.0f;
#pragma unroll
    for (int e = 0; e < 8; ++e) {
      const int m = m0 + 8 * g + e;
      float v = acc[t][e] + bv;
      if (flags & 1) v = v > 0.0f ? v : 0.0f;
      if (flags & 2) {
        // V-transpose store: vT[(b*H + h)*hd + d][s], m = b*SEQ+s, n = h*hd+d
        const int b_ = m >> 10, s = m & 1023;
        const int h_ = n >> 6, d = n & 63;
        outH[((size_t)(b_ * NHEAD + h_) * HDIM + d) * SEQ + s] = (_Float16)v;
      } else {
        if (outF) outF[(size_t)m * N + n] = v;
        if (outH) outH[(size_t)m * N + n] = (_Float16)v;
      }
    }
  }
}

// ---------------------------------------------------------------------------
// Attention: per wave -> 16 query rows of one (b,h).
//   scores = (q . k^T)/8 via WMMA into LDS (16 x 1024 fp32, padded stride),
//   single-pass row softmax (shuffle reductions), normalized weights written
//   straight to d_out (fp32), then ctx = P @ V via WMMA reading P from LDS
//   and V from the transposed f16 vT buffer (contiguous, pipelined B-frags).
// Block = 64 threads (2 waves). LDS = 2 * 16 * 1028 * 4 = 131584 B.
// ---------------------------------------------------------------------------
#define ATT_WAVES 2
#define SSTRIDE   1028   // 16B-aligned, breaks the 8-row bank collision

__global__ __launch_bounds__(64) void attn_kernel(
    const _Float16* __restrict__ q, const _Float16* __restrict__ k,
    const _Float16* __restrict__ vT, float* __restrict__ attn_out,
    _Float16* __restrict__ ctx) {
  __shared__ float sm[ATT_WAVES][16][SSTRIDE];
  __shared__ float sinv[ATT_WAVES][16];

  const int lane = threadIdx.x & 31;
  const int wave = threadIdx.x >> 5;
  const int g = lane >> 4;
  const int r = lane & 15;
  const int bh = blockIdx.y;            // b*NHEAD + h
  const int b_ = bh >> 3, h_ = bh & 7;
  const int l0 = blockIdx.x * (16 * ATT_WAVES) + wave * 16;

  float (*S_)[SSTRIDE] = sm[wave];

  // Q A-frags for the two K=32 halves of hd=64 (held across the whole s loop)
  v16h aq[2];
  {
    const _Float16* qrow =
        q + (size_t)(b_ * SEQ + l0 + r) * DMODEL + h_ * HDIM + 16 * g;
#pragma unroll
    for (int c = 0; c < 2; ++c)
      aq[c] = cat8(*(const v8h*)(qrow + 32 * c),
                   *(const v8h*)(qrow + 32 * c + 8));
  }

  // ---- scores -> LDS (K-fragment loads pipelined one s-tile ahead) ----
  const float scale = 0.125f;  // 1/sqrt(64)
  const _Float16* krow0 =
      k + (size_t)(b_ * SEQ + r) * DMODEL + h_ * HDIM + 16 * g;
  v16h bk0 = *(const v16h*)(krow0);
  v16h bk1 = *(const v16h*)(krow0 + 32);
#pragma unroll 4
  for (int s0 = 0; s0 < SEQ; s0 += 16) {
    const int sn = (s0 + 16) & (SEQ - 1);
    const _Float16* kr = krow0 + (size_t)sn * DMODEL;
    const v16h bk0n = *(const v16h*)(kr);
    const v16h bk1n = *(const v16h*)(kr + 32);

    v8f acc = {};
    acc = __builtin_amdgcn_wmma_f32_16x16x32_f16(
        false, aq[0], false, bk0, (short)0, acc, false, false);
    acc = __builtin_amdgcn_wmma_f32_16x16x32_f16(
        false, aq[1], false, bk1, (short)0, acc, false, false);
#pragma unroll
    for (int e = 0; e < 8; ++e) S_[8 * g + e][s0 + r] = acc[e] * scale;
    bk0 = bk0n;
    bk1 = bk1n;
  }
  __syncthreads();

  // ---- softmax per row; write normalized weights to global (fp32) ----
  float* wout_base = attn_out + ((size_t)bh * SEQ + l0) * SEQ;
  for (int row = 0; row < 16; ++row) {
    float mx = -3.4e38f;
    for (int c = lane; c < SEQ; c += 32) mx = fmaxf(mx, S_[row][c]);
    mx = wred_max(mx);
    float sum = 0.0f;
    for (int c = lane; c < SEQ; c += 32) {
      float e = __expf(S_[row][c] - mx);
      S_[row][c] = e;       // keep unnormalized exp for the PV matmul
      sum += e;
    }
    sum = wred_sum(sum);
    const float iv = 1.0f / sum;
    if (lane == 0) sinv[wave][row] = iv;
    float* dst = wout_base + (size_t)row * SEQ;
    for (int c = lane; c < SEQ; c += 32) dst[c] = S_[row][c] * iv;
  }

  // ---- ctx = P @ V (K = S = 1024); V frags pipelined one K-step ahead ----
  v8f cacc[4] = {};
  const _Float16* vrow0 = vT + ((size_t)bh * HDIM + r) * SEQ + 16 * g;
  v16h bv[4];
#pragma unroll
  for (int t = 0; t < 4; ++t)
    bv[t] = *(const v16h*)(vrow0 + (size_t)(16 * t) * SEQ);
#pragma unroll 4
  for (int s0 = 0; s0 < SEQ; s0 += 32) {
    const int sn = (s0 + 32) & (SEQ - 1);
    v16h bvn[4];
#pragma unroll
    for (int t = 0; t < 4; ++t)
      bvn[t] = *(const v16h*)(vrow0 + (size_t)(16 * t) * SEQ + sn);

    const float* prow = &S_[r][s0 + 16 * g];
    const v16h a =
        pack16(ld4(prow), ld4(prow + 4), ld4(prow + 8), ld4(prow + 12));
#pragma unroll
    for (int t = 0; t < 4; ++t)
      cacc[t] = __builtin_amdgcn_wmma_f32_16x16x32_f16(
          false, a, false, bv[t], (short)0, cacc[t], false, false);
#pragma unroll
    for (int t = 0; t < 4; ++t) bv[t] = bvn[t];
  }
#pragma unroll
  for (int t = 0; t < 4; ++t)
#pragma unroll
    for (int e = 0; e < 8; ++e) {
      const int row = 8 * g + e;
      ctx[(size_t)(b_ * SEQ + l0 + row) * DMODEL + h_ * HDIM + 16 * t + r] =
          (_Float16)(cacc[t][e] * sinv[wave][row]);
    }
}

// ---------------------------------------------------------------------------
// out[row,:] = LayerNorm(x[row,:] + y[row,:]) * gamma + beta   (row len 512)
// Optional f16 duplicate (feeds the next GEMM's A operand).
// Block = 256 threads (8 waves), one row per wave.
// ---------------------------------------------------------------------------
__global__ __launch_bounds__(256) void add_ln_kernel(
    const float* __restrict__ x, const float* __restrict__ y,
    const float* __restrict__ gamma, const float* __restrict__ beta,
    float* __restrict__ out, _Float16* __restrict__ outH) {
  const int lane = threadIdx.x & 31;
  const int wave = threadIdx.x >> 5;
  const size_t row = (size_t)blockIdx.x * 8 + wave;
  const float* xr = x + row * DMODEL;
  const float* yr = y + row * DMODEL;
  float vv[16];
  float sum = 0.0f;
#pragma unroll
  for (int i = 0; i < 16; ++i) {
    const int c = lane + 32 * i;
    vv[i] = xr[c] + yr[c];
    sum += vv[i];
  }
  sum = wred_sum(sum);
  const float mu = sum * (1.0f / DMODEL);
  float var = 0.0f;
#pragma unroll
  for (int i = 0; i < 16; ++i) {
    const float d = vv[i] - mu;
    var += d * d;
  }
  var = wred_sum(var) * (1.0f / DMODEL);
  const float rs = rsqrtf(var + 1e-5f);
  float* outr = out + row * DMODEL;
  _Float16* outhr = outH ? outH + row * DMODEL : nullptr;
#pragma unroll
  for (int i = 0; i < 16; ++i) {
    const int c = lane + 32 * i;
    const float o = (vv[i] - mu) * rs * gamma[c] + beta[c];
    outr[c] = o;
    if (outhr) outhr[c] = (_Float16)o;
  }
}

// ---------------------------------------------------------------------------
extern "C" void kernel_launch(void* const* d_in, const int* in_sizes, int n_in,
                              void* d_out, int out_size, void* d_ws,
                              size_t ws_size, hipStream_t stream) {
  (void)in_sizes; (void)n_in; (void)out_size; (void)ws_size;
  const float* x1   = (const float*)d_in[0];
  const float* x2   = (const float*)d_in[1];
  const float* win1 = (const float*)d_in[2];
  const float* bin1 = (const float*)d_in[3];
  const float* wout1= (const float*)d_in[4];
  const float* bout1= (const float*)d_in[5];
  const float* win2 = (const float*)d_in[6];
  const float* bin2 = (const float*)d_in[7];
  const float* wout2= (const float*)d_in[8];
  const float* bout2= (const float*)d_in[9];
  const float* g1   = (const float*)d_in[10];
  const float* b1   = (const float*)d_in[11];
  const float* g2   = (const float*)d_in[12];
  const float* b2   = (const float*)d_in[13];
  const float* fw1  = (const float*)d_in[14];
  const float* fb1  = (const float*)d_in[15];
  const float* fw2  = (const float*)d_in[16];
  const float* fb2  = (const float*)d_in[17];
  const float* g3   = (const float*)d_in[18];
  const float* b3   = (const float*)d_in[19];

  float* out = (float*)d_out;
  float* ws  = (float*)d_ws;

  const size_t TD = (size_t)NTOK * DMODEL;  // 4,194,304 elements

  // fp32 scratch
  float* o   = ws;            // GEMM fp32 results feeding residual+LN
  float* x1n = ws + TD;
  float* x2n = ws + 2 * TD;

  // f16 scratch
  _Float16* hb   = (_Float16*)(ws + 3 * TD);
  _Float16* x1h  = hb;
  _Float16* x2h  = hb + TD;
  _Float16* qh   = hb + 2 * TD;
  _Float16* kh   = hb + 3 * TD;
  _Float16* vTh  = hb + 4 * TD;
  _Float16* ctxh = hb + 5 * TD;
  _Float16* x1nh = hb + 6 * TD;
  _Float16* x2nh = hb + 7 * TD;
  _Float16* ffh  = hb + 8 * TD;            // NTOK x 1024 (2*TD)
  _Float16* win1h = hb + 10 * TD;          // 3D*D = 786,432
  _Float16* win2h = win1h + 3 * DMODEL * DMODEL;
  _Float16* wout1h = win2h + 3 * DMODEL * DMODEL;
  _Float16* wout2h = wout1h + DMODEL * DMODEL;
  _Float16* fw1h  = wout2h + DMODEL * DMODEL;   // 1024*512
  _Float16* fw2h  = fw1h + 1024 * DMODEL;       // 512*1024

  float* w1to2 = out + 2 * TD;                            // (B,H,L,S)
  float* w2to1 = out + 2 * TD + (size_t)64 * SEQ * SEQ;   // 64 = B*H

  const dim3 blk(256);
  const dim3 g512(DMODEL / 64, NTOK / 128);
  const dim3 g1024(1024 / 64, NTOK / 128);
  const dim3 gattn(SEQ / (16 * ATT_WAVES), 8 * NHEAD);
  const dim3 battn(32 * ATT_WAVES);
  const dim3 gln(NTOK / 8);
  auto cvtg = [](size_t n) { return dim3((unsigned)((n / 8 + 255) / 256)); };

  // ---- one-time f16 staging (weights + inputs) ----
  cvt_kernel<<<cvtg(TD), blk, 0, stream>>>(x1, x1h, (int)TD);
  cvt_kernel<<<cvtg(TD), blk, 0, stream>>>(x2, x2h, (int)TD);
  cvt_kernel<<<cvtg(3 * DMODEL * DMODEL), blk, 0, stream>>>(win1, win1h, 3 * DMODEL * DMODEL);
  cvt_kernel<<<cvtg(3 * DMODEL * DMODEL), blk, 0, stream>>>(win2, win2h, 3 * DMODEL * DMODEL);
  cvt_kernel<<<cvtg(DMODEL * DMODEL), blk, 0, stream>>>(wout1, wout1h, DMODEL * DMODEL);
  cvt_kernel<<<cvtg(DMODEL * DMODEL), blk, 0, stream>>>(wout2, wout2h, DMODEL * DMODEL);
  cvt_kernel<<<cvtg(1024 * DMODEL), blk, 0, stream>>>(fw1, fw1h, 1024 * DMODEL);
  cvt_kernel<<<cvtg(1024 * DMODEL), blk, 0, stream>>>(fw2, fw2h, 1024 * DMODEL);

  // ---- MHA 1: q from x1, k/v from x2 ----
  gemm_kernel<DMODEL><<<g512, blk, 0, stream>>>(x1h, win1h,              bin1,        nullptr, qh,  DMODEL, 0);
  gemm_kernel<DMODEL><<<g512, blk, 0, stream>>>(x2h, win1h + 512 * 512,  bin1 + 512,  nullptr, kh,  DMODEL, 0);
  gemm_kernel<DMODEL><<<g512, blk, 0, stream>>>(x2h, win1h + 1024 * 512, bin1 + 1024, nullptr, vTh, DMODEL, 2);
  attn_kernel<<<gattn, battn, 0, stream>>>(qh, kh, vTh, w1to2, ctxh);
  gemm_kernel<DMODEL><<<g512, blk, 0, stream>>>(ctxh, wout1h, bout1, o, nullptr, DMODEL, 0);
  add_ln_kernel<<<gln, blk, 0, stream>>>(x1, o, g1, b1, x1n, x1nh);

  // ---- MHA 2: q from x2, k/v from updated x1 ----
  gemm_kernel<DMODEL><<<g512, blk, 0, stream>>>(x2h,  win2h,              bin2,        nullptr, qh,  DMODEL, 0);
  gemm_kernel<DMODEL><<<g512, blk, 0, stream>>>(x1nh, win2h + 512 * 512,  bin2 + 512,  nullptr, kh,  DMODEL, 0);
  gemm_kernel<DMODEL><<<g512, blk, 0, stream>>>(x1nh, win2h + 1024 * 512, bin2 + 1024, nullptr, vTh, DMODEL, 2);
  attn_kernel<<<gattn, battn, 0, stream>>>(qh, kh, vTh, w2to1, ctxh);
  gemm_kernel<DMODEL><<<g512, blk, 0, stream>>>(ctxh, wout2h, bout2, o, nullptr, DMODEL, 0);
  add_ln_kernel<<<gln, blk, 0, stream>>>(x2, o, g2, b2, x2n, x2nh);

  // ---- FFN on x1n -> out[0 : TD) ----
  gemm_kernel<DMODEL><<<g1024, blk, 0, stream>>>(x1nh, fw1h, fb1, nullptr, ffh, 1024, 1);
  gemm_kernel<1024><<<g512,  blk, 0, stream>>>(ffh, fw2h, fb2, o, nullptr, DMODEL, 0);
  add_ln_kernel<<<gln, blk, 0, stream>>>(x1n, o, g3, b3, out, nullptr);

  // ---- FFN on x2n -> out[TD : 2*TD) ----
  gemm_kernel<DMODEL><<<g1024, blk, 0, stream>>>(x2nh, fw1h, fb1, nullptr, ffh, 1024, 1);
  gemm_kernel<1024><<<g512,  blk, 0, stream>>>(ffh, fw2h, fb2, o, nullptr, DMODEL, 0);
  add_ln_kernel<<<gln, blk, 0, stream>>>(x2n, o, g3, b3, out + TD, nullptr);
}
